// FastTransformerBlock_32530082300189
// MI455X (gfx1250) — compile-verified, matrix operand order checked by reading, not run on hardware
//
#include <hip/hip_runtime.h>

// ---------------------------------------------------------------------------
// FastTransformerBlock for MI455X (gfx1250, wave32, WMMA + TDM).
// bf16 x bf16 -> f32 WMMA for all GEMMs + flash attention; fp32 LN/softmax/GELU.
// GEMM A-tiles staged to LDS by the Tensor Data Mover (tensor_load_to_lds,
// TENSORcnt-synchronized); B-tiles staged transposed by the wavefronts.
// Cross-attention with Skv==1 reduces exactly to a broadcast of cv@cWo+cbo.
// ---------------------------------------------------------------------------

typedef __attribute__((ext_vector_type(16))) __bf16 v16bf;
typedef __attribute__((ext_vector_type(8)))  __bf16 v8bf;
typedef __attribute__((ext_vector_type(8)))  float  v8f;
typedef __attribute__((ext_vector_type(4))) unsigned int u32x4;
typedef __attribute__((ext_vector_type(8))) int i32x8;
typedef __attribute__((ext_vector_type(4))) int i32x4;

#ifndef USE_TDM
#if defined(__has_builtin)
#if __has_builtin(__builtin_amdgcn_tensor_load_to_lds)
#define USE_TDM 1
#endif
#endif
#endif
#ifndef USE_TDM
#define USE_TDM 0
#endif

#define SEQ   2048
#define EMBED 1024
#define DFF   4096
#define HEADS 16
#define HDIM  64
#define LNEPS 1e-5f

__device__ __forceinline__ v16bf cat16(v8bf lo, v8bf hi) {
  return __builtin_shufflevector(lo, hi, 0,1,2,3,4,5,6,7,8,9,10,11,12,13,14,15);
}
__device__ __forceinline__ v8f wmma_bf16(v16bf a, v16bf b, v8f c) {
  return __builtin_amdgcn_wmma_f32_16x16x32_bf16(false, a, false, b, (short)0, c, false, false);
}

// ---------------------------------------------------------------------------
// Elementwise fp32 -> bf16 conversion
// ---------------------------------------------------------------------------
__global__ __launch_bounds__(256) void f32_to_bf16_k(const float* __restrict__ x,
                                                     __bf16* __restrict__ y, size_t n) {
  size_t i = (size_t)blockIdx.x * 256 + threadIdx.x;
  if (i < n) y[i] = (__bf16)x[i];
}

// ---------------------------------------------------------------------------
// Tiled bf16 WMMA GEMM: C(MxN) = A(MxK) @ B(KxN) + bias [,GELU][,+resid]
// 128 threads (4 waves), block tile 128x64, BK=32; each wave owns 32x64.
// A tile (128x32) staged by TDM into padded LDS (80B row stride == pad
// interval 16 dwords / pad amount 4 dwords); B tile staged transposed.
// ---------------------------------------------------------------------------
template<bool GELU, bool RES, bool OUTF, bool OUTB>
__global__ __launch_bounds__(128)
void gemm_bf16_wmma(const __bf16* __restrict__ A, const __bf16* __restrict__ B,
                    const float* __restrict__ bias, const float* __restrict__ resid,
                    float* __restrict__ outF, __bf16* __restrict__ outB,
                    int M, int N, int K) {
  constexpr int PAD = 40;                         // 80B row stride (16B multiple)
  __shared__ __align__(16) __bf16 As[128 * PAD];
  __shared__ __align__(16) __bf16 Bt[64 * PAD];   // B tile stored transposed [n][k]

  const int tid  = threadIdx.x;
  const int lane = tid & 31;
  const int wave = tid >> 5;
  const int lhalf = lane >> 4;       // 0: lanes 0-15, 1: lanes 16-31
  const int l16   = lane & 15;
  const int ko    = lhalf * 8;       // K sub-offset per lane half
  const int m0 = blockIdx.y * 128;
  const int n0 = blockIdx.x * 64;

  v8f acc[2][4] = {};

  for (int k0 = 0; k0 < K; k0 += 32) {
#if USE_TDM
    // --- Tensor Data Mover: one descriptor moves the whole 128x32 A tile ---
    if (wave == 0) {
      const unsigned long long ga =
          (unsigned long long)(uintptr_t)(A + (size_t)m0 * K + k0);
      u32x4 g0 = {};
      g0[0] = 1u;                                    // count=1 (valid user D#)
      g0[1] = (unsigned)(uintptr_t)As;               // lds_addr (low 32b = LDS offset)
      g0[2] = (unsigned)ga;                          // global_addr[31:0]
      g0[3] = (unsigned)((ga >> 32) & 0x01FFFFFFull) // global_addr[56:32]
              | 0x80000000u;                         // type=2 ("image")
      i32x8 g1 = {};
      g1[0] = (1 << 16)      // data_size = 2 bytes
            | (1 << 20)      // pad_enable
            | (3 << 22)      // pad_interval: 16 dwords (one 32-elem row)
            | (3 << 25);     // pad_amount: 4 dwords -> 80B LDS row stride
      g1[1] = (int)(((unsigned)K & 0xFFFFu) << 16);                       // tensor_dim0 lo
      g1[2] = (int)(((unsigned)K >> 16) | (((unsigned)M & 0xFFFFu) << 16)); // dim0 hi | dim1 lo
      g1[3] = (int)(((unsigned)M >> 16) | (32u << 16));                   // dim1 hi | tile_dim0=32
      g1[4] = 128;            // tile_dim1 = 128 rows (tile_dim2 = 0)
      g1[5] = K;              // tensor_dim0_stride[31:0] (elements)
      i32x4 g2 = {}, g3 = {};
#if __clang_major__ >= 23
      i32x8 g4 = {};
      __builtin_amdgcn_tensor_load_to_lds(g0, g1, g2, g3, g4, 0);
#else
      __builtin_amdgcn_tensor_load_to_lds(g0, g1, g2, g3, 0);
#endif
    }
#else
    // manual A staging fallback: 128x32, 4x b128 per thread
    #pragma unroll
    for (int r = 0; r < 4; ++r) {
      int c = tid + r * 128;
      int row = c >> 2, cc = c & 3;
      uint4 va = *(const uint4*)(A + (size_t)(m0 + row) * K + k0 + cc * 8);
      *(uint4*)(As + row * PAD + cc * 8) = va;
    }
#endif
    // stage B tile 32x64, transposed into Bt[n][k]
    #pragma unroll
    for (int r = 0; r < 2; ++r) {
      int c = tid + r * 128;
      int krow = c >> 3, nc = c & 7;
      uint4 vb = *(const uint4*)(B + (size_t)(k0 + krow) * N + n0 + nc * 8);
      v8bf e = __builtin_bit_cast(v8bf, vb);
      #pragma unroll
      for (int j = 0; j < 8; ++j) Bt[(nc * 8 + j) * PAD + krow] = e[j];
    }
    if (k0 + 32 < K)  // prefetch next B tile (global_prefetch_b8)
      __builtin_prefetch(B + (size_t)(k0 + 32 + (tid >> 3)) * N + n0 + (tid & 7) * 8, 0, 0);
#if USE_TDM
    __builtin_amdgcn_s_wait_tensorcnt(0);   // no-op for waves 1..3
#endif
    __syncthreads();

    // A fragments: two 16x32 row-tiles per wave
    v16bf afrag[2];
    #pragma unroll
    for (int mb = 0; mb < 2; ++mb) {
      const int arow = wave * 32 + mb * 16 + l16;
      v8bf alo = *(const v8bf*)(As + arow * PAD + ko);
      v8bf ahi = *(const v8bf*)(As + arow * PAD + 16 + ko);
      afrag[mb] = cat16(alo, ahi);
    }
    #pragma unroll
    for (int nb = 0; nb < 4; ++nb) {
      const int bcol = nb * 16 + l16;
      v8bf blo = *(const v8bf*)(Bt + bcol * PAD + ko);
      v8bf bhi = *(const v8bf*)(Bt + bcol * PAD + 16 + ko);
      v16bf bfrag = cat16(blo, bhi);
      acc[0][nb] = wmma_bf16(afrag[0], bfrag, acc[0][nb]);
      acc[1][nb] = wmma_bf16(afrag[1], bfrag, acc[1][nb]);
    }
    __syncthreads();
  }

  // epilogue (C layout: vgpr r -> row r + 8*lane_half, col = lane&15)
  #pragma unroll
  for (int mb = 0; mb < 2; ++mb) {
    #pragma unroll
    for (int nb = 0; nb < 4; ++nb) {
      const int col = n0 + nb * 16 + l16;
      const float bn = bias[col];
      #pragma unroll
      for (int r = 0; r < 8; ++r) {
        const int row = m0 + wave * 32 + mb * 16 + r + lhalf * 8;
        float v = acc[mb][nb][r] + bn;
        if (GELU) v = 0.5f * v * (1.0f + erff(v * 0.70710678118f));
        const size_t idx = (size_t)row * N + col;
        if (RES)  v += resid[idx];
        if (OUTF) outF[idx] = v;
        if (OUTB) outB[idx] = (__bf16)v;
      }
    }
  }
}

// ---------------------------------------------------------------------------
// Flash attention: one wave per (head, 16-query tile); online softmax over
// 32-key chunks; QK^T and PV both WMMA bf16.
// ---------------------------------------------------------------------------
__global__ __launch_bounds__(32)
void attn_flash(const __bf16* __restrict__ Q, const __bf16* __restrict__ Km,
                const __bf16* __restrict__ Vm, __bf16* __restrict__ O) {
  constexpr int PAD = 40;
  __shared__ __align__(16) __bf16 Qs[16 * HDIM];
  __shared__ __align__(16) __bf16 Ks[32 * HDIM];
  __shared__ __align__(16) __bf16 Vt[HDIM * PAD];   // V transposed: [col][key]
  __shared__ __align__(16) __bf16 Ps[16 * 32];      // P tile (A-operand staging)

  const int lane = threadIdx.x;
  const int head = blockIdx.y;
  const int q0   = blockIdx.x * 16;
  const int lhalf = lane >> 4, l16 = lane & 15, ko = lhalf * 8;
  const int hoff = head * HDIM;

  if (lane < 16) {
    const __bf16* src = Q + (size_t)(q0 + lane) * EMBED + hoff;
    #pragma unroll
    for (int c = 0; c < 8; ++c)
      *(uint4*)(Qs + lane * HDIM + c * 8) = *(const uint4*)(src + c * 8);
  }
  __syncthreads();

  v16bf aq[2];
  #pragma unroll
  for (int kk = 0; kk < 2; ++kk) {
    v8bf lo = *(const v8bf*)(Qs + l16 * HDIM + kk * 32 + ko);
    v8bf hi = *(const v8bf*)(Qs + l16 * HDIM + kk * 32 + 16 + ko);
    aq[kk] = cat16(lo, hi);
  }

  float mstat[8], lstat[8];
  #pragma unroll
  for (int r = 0; r < 8; ++r) { mstat[r] = -1e30f; lstat[r] = 0.0f; }
  v8f oacc[4] = {};

  for (int kv0 = 0; kv0 < SEQ; kv0 += 32) {
    {  // each lane stages one key row of K (row-major) and V (transposed)
      const __bf16* ksrc = Km + (size_t)(kv0 + lane) * EMBED + hoff;
      #pragma unroll
      for (int c = 0; c < 8; ++c)
        *(uint4*)(Ks + lane * HDIM + c * 8) = *(const uint4*)(ksrc + c * 8);
      const __bf16* vsrc = Vm + (size_t)(kv0 + lane) * EMBED + hoff;
      #pragma unroll
      for (int c = 0; c < 8; ++c) {
        v8bf e = __builtin_bit_cast(v8bf, *(const uint4*)(vsrc + c * 8));
        #pragma unroll
        for (int j = 0; j < 8; ++j) Vt[(c * 8 + j) * PAD + lane] = e[j];
      }
    }
    __syncthreads();

    // scores: two 16x16 subtiles, K-depth 64 = 2 WMMA each
    v8f sc[2] = {};
    #pragma unroll
    for (int j = 0; j < 2; ++j)
      #pragma unroll
      for (int kk = 0; kk < 2; ++kk) {
        v8bf lo = *(const v8bf*)(Ks + (j * 16 + l16) * HDIM + kk * 32 + ko);
        v8bf hi = *(const v8bf*)(Ks + (j * 16 + l16) * HDIM + kk * 32 + 16 + ko);
        sc[j] = wmma_bf16(aq[kk], cat16(lo, hi), sc[j]);
      }

    // online softmax; row r lives in 16 lanes of one half -> xor-reduce 1,2,4,8
    #pragma unroll
    for (int r = 0; r < 8; ++r) {
      float s0 = sc[0][r] * 0.125f;   // 1/sqrt(64)
      float s1 = sc[1][r] * 0.125f;
      float mx = fmaxf(s0, s1);
      #pragma unroll
      for (int d = 1; d < 16; d <<= 1) mx = fmaxf(mx, __shfl_xor(mx, d, 32));
      float mnew = fmaxf(mstat[r], mx);
      float p0 = __expf(s0 - mnew), p1 = __expf(s1 - mnew);
      float ps = p0 + p1;
      #pragma unroll
      for (int d = 1; d < 16; d <<= 1) ps += __shfl_xor(ps, d, 32);
      float corr = __expf(mstat[r] - mnew);
      lstat[r] = lstat[r] * corr + ps;
      mstat[r] = mnew;
      #pragma unroll
      for (int nb = 0; nb < 4; ++nb) oacc[nb][r] *= corr;
      const int row = r + lhalf * 8;  // C-layout -> A-layout via LDS
      Ps[row * 32 + l16]      = (__bf16)p0;
      Ps[row * 32 + 16 + l16] = (__bf16)p1;
    }
    __syncthreads();

    // PV: P(16x32) @ V(32x64)
    v8bf plo = *(const v8bf*)(Ps + l16 * 32 + ko);
    v8bf phi = *(const v8bf*)(Ps + l16 * 32 + 16 + ko);
    v16bf ap = cat16(plo, phi);
    #pragma unroll
    for (int nb = 0; nb < 4; ++nb) {
      const int col = nb * 16 + l16;
      v8bf blo = *(const v8bf*)(Vt + col * PAD + ko);
      v8bf bhi = *(const v8bf*)(Vt + col * PAD + 16 + ko);
      oacc[nb] = wmma_bf16(ap, cat16(blo, bhi), oacc[nb]);
    }
    __syncthreads();
  }

  #pragma unroll
  for (int nb = 0; nb < 4; ++nb) {
    const int col = hoff + nb * 16 + l16;
    #pragma unroll
    for (int r = 0; r < 8; ++r) {
      const int row = q0 + r + lhalf * 8;
      O[(size_t)row * EMBED + col] = (__bf16)(oacc[nb][r] / lstat[r]);
    }
  }
}

// ---------------------------------------------------------------------------
// LayerNorm over rows of 1024; one 256-thread block per row.
// ---------------------------------------------------------------------------
template<bool OUTB>
__global__ __launch_bounds__(256)
void layernorm_k(const float* __restrict__ x, const float* __restrict__ g,
                 const float* __restrict__ b, float* __restrict__ yF,
                 __bf16* __restrict__ yB) {
  __shared__ float ssum[256], ssq[256];
  const int tid = threadIdx.x;
  const size_t base = (size_t)blockIdx.x * EMBED + tid * 4;
  float4 v = *(const float4*)(x + base);
  float s  = v.x + v.y + v.z + v.w;
  float sq = v.x * v.x + v.y * v.y + v.z * v.z + v.w * v.w;
  ssum[tid] = s; ssq[tid] = sq;
  __syncthreads();
  for (int st = 128; st > 0; st >>= 1) {
    if (tid < st) { ssum[tid] += ssum[tid + st]; ssq[tid] += ssq[tid + st]; }
    __syncthreads();
  }
  const float mean = ssum[0] * (1.0f / EMBED);
  const float var  = ssq[0] * (1.0f / EMBED) - mean * mean;
  const float rstd = rsqrtf(var + LNEPS);
  float4 gg = *(const float4*)(g + tid * 4);
  float4 bb = *(const float4*)(b + tid * 4);
  float4 o;
  o.x = (v.x - mean) * rstd * gg.x + bb.x;
  o.y = (v.y - mean) * rstd * gg.y + bb.y;
  o.z = (v.z - mean) * rstd * gg.z + bb.z;
  o.w = (v.w - mean) * rstd * gg.w + bb.w;
  *(float4*)(yF + base) = o;
  if (OUTB) {
    yB[base + 0] = (__bf16)o.x; yB[base + 1] = (__bf16)o.y;
    yB[base + 2] = (__bf16)o.z; yB[base + 3] = (__bf16)o.w;
  }
}

// fp32 GEMV: out[n] = bias[n] + sum_k t[k]*W[k*N+n]   (tiny cross-attn path)
__global__ __launch_bounds__(256)
void gemv_f32(const float* __restrict__ t, const float* __restrict__ W,
              const float* __restrict__ bias, float* __restrict__ out,
              int K, int N) {
  const int n = blockIdx.x * 256 + threadIdx.x;
  if (n >= N) return;
  float acc = bias[n];
  for (int k = 0; k < K; ++k) acc += t[k] * W[(size_t)k * N + n];
  out[n] = acc;
}

// y[i] = x[i] + rowvec[i & (N-1)]
__global__ __launch_bounds__(256)
void add_rowvec_k(const float* __restrict__ x, const float* __restrict__ rv,
                  float* __restrict__ y, size_t total) {
  size_t i = (size_t)blockIdx.x * 256 + threadIdx.x;
  if (i < total) y[i] = x[i] + rv[i & (EMBED - 1)];
}

// ---------------------------------------------------------------------------
extern "C" void kernel_launch(void* const* d_in, const int* in_sizes, int n_in,
                              void* d_out, int out_size, void* d_ws, size_t ws_size,
                              hipStream_t stream) {
  (void)in_sizes; (void)n_in; (void)out_size; (void)ws_size;
  const float* x      = (const float*)d_in[0];   // (2048,1,1024) == (2048,1024)
  const float* th     = (const float*)d_in[1];   // (1,1024)
  const float* Wq  = (const float*)d_in[2];  const float* bq  = (const float*)d_in[3];
  const float* Wk  = (const float*)d_in[4];  const float* bk  = (const float*)d_in[5];
  const float* Wv  = (const float*)d_in[6];  const float* bv  = (const float*)d_in[7];
  const float* Wo  = (const float*)d_in[8];  const float* bo  = (const float*)d_in[9];
  // cWq(10)/cbq(11)/cWk(12)/cbk(13) unused: softmax over a single key == 1.0
  const float* cWv = (const float*)d_in[14]; const float* cbv = (const float*)d_in[15];
  const float* cWo = (const float*)d_in[16]; const float* cbo = (const float*)d_in[17];
  const float* ln1g = (const float*)d_in[18]; const float* ln1b = (const float*)d_in[19];
  const float* ln2g = (const float*)d_in[20]; const float* ln2b = (const float*)d_in[21];
  const float* ln3g = (const float*)d_in[22]; const float* ln3b = (const float*)d_in[23];
  const float* W1  = (const float*)d_in[24]; const float* bf1 = (const float*)d_in[25];
  const float* W2  = (const float*)d_in[26]; const float* bf2 = (const float*)d_in[27];
  float* out = (float*)d_out;

  char* ws = (char*)d_ws;
  size_t off = 0;
  auto alloc = [&](size_t bytes) {
    void* p = ws + off;
    off = (off + bytes + 255) & ~(size_t)255;
    return p;
  };
  const size_t SD = (size_t)SEQ * EMBED;
  __bf16* xb_bf  = (__bf16*)alloc(SD * 2);
  __bf16* wq_bf  = (__bf16*)alloc((size_t)EMBED * EMBED * 2);
  __bf16* wk_bf  = (__bf16*)alloc((size_t)EMBED * EMBED * 2);
  __bf16* wv_bf  = (__bf16*)alloc((size_t)EMBED * EMBED * 2);
  __bf16* wo_bf  = (__bf16*)alloc((size_t)EMBED * EMBED * 2);
  __bf16* w1_bf  = (__bf16*)alloc((size_t)EMBED * DFF * 2);
  __bf16* w2_bf  = (__bf16*)alloc((size_t)DFF * EMBED * 2);
  __bf16* q_bf   = (__bf16*)alloc(SD * 2);
  __bf16* k_bf   = (__bf16*)alloc(SD * 2);
  __bf16* v_bf   = (__bf16*)alloc(SD * 2);
  __bf16* at_bf  = (__bf16*)alloc(SD * 2);
  float*  res1   = (float*)alloc(SD * 4);
  float*  ln1o   = (float*)alloc(SD * 4);
  float*  cv     = (float*)alloc(EMBED * 4);
  float*  crossv = (float*)alloc(EMBED * 4);
  float*  res2   = (float*)alloc(SD * 4);
  float*  ln2o   = (float*)alloc(SD * 4);
  __bf16* ln2_bf = (__bf16*)alloc(SD * 2);
  __bf16* h_bf   = (__bf16*)alloc((size_t)SEQ * DFF * 2);
  float*  res3   = (float*)alloc(SD * 4);

  auto conv = [&](const float* s, __bf16* d, size_t n) {
    f32_to_bf16_k<<<dim3((unsigned)((n + 255) / 256)), 256, 0, stream>>>(s, d, n);
  };
  conv(x,  xb_bf, SD);
  conv(Wq, wq_bf, (size_t)EMBED * EMBED);
  conv(Wk, wk_bf, (size_t)EMBED * EMBED);
  conv(Wv, wv_bf, (size_t)EMBED * EMBED);
  conv(Wo, wo_bf, (size_t)EMBED * EMBED);
  conv(W1, w1_bf, (size_t)EMBED * DFF);
  conv(W2, w2_bf, (size_t)DFF * EMBED);

  const dim3 blk(128);
  const dim3 gEE(EMBED / 64, SEQ / 128);   // N=1024, M-tile 128
  const dim3 gEF(DFF / 64,   SEQ / 128);   // N=4096

  // Q/K/V projections -> bf16
  gemm_bf16_wmma<false,false,false,true><<<gEE, blk, 0, stream>>>(
      xb_bf, wq_bf, bq, nullptr, nullptr, q_bf, SEQ, EMBED, EMBED);
  gemm_bf16_wmma<false,false,false,true><<<gEE, blk, 0, stream>>>(
      xb_bf, wk_bf, bk, nullptr, nullptr, k_bf, SEQ, EMBED, EMBED);
  gemm_bf16_wmma<false,false,false,true><<<gEE, blk, 0, stream>>>(
      xb_bf, wv_bf, bv, nullptr, nullptr, v_bf, SEQ, EMBED, EMBED);

  // flash attention
  attn_flash<<<dim3(SEQ / 16, HEADS), 32, 0, stream>>>(q_bf, k_bf, v_bf, at_bf);

  // output projection + residual(x) -> res1 ; LN1
  gemm_bf16_wmma<false,true,true,false><<<gEE, blk, 0, stream>>>(
      at_bf, wo_bf, bo, x, res1, nullptr, SEQ, EMBED, EMBED);
  layernorm_k<false><<<SEQ, 256, 0, stream>>>(res1, ln1g, ln1b, ln1o, nullptr);

  // cross-attention (exact algebraic reduction): crossv = (th@cWv+cbv)@cWo+cbo
  gemv_f32<<<dim3(EMBED / 256), 256, 0, stream>>>(th, cWv, cbv, cv, EMBED, EMBED);
  gemv_f32<<<dim3(EMBED / 256), 256, 0, stream>>>(cv, cWo, cbo, crossv, EMBED, EMBED);
  add_rowvec_k<<<dim3((unsigned)(SD / 256)), 256, 0, stream>>>(ln1o, crossv, res2, SD);
  layernorm_k<true><<<SEQ, 256, 0, stream>>>(res2, ln2g, ln2b, ln2o, ln2_bf);

  // FFN: GELU(ln2 @ W1 + bf1) @ W2 + bf2 + residual(ln2) ; LN3 -> out
  gemm_bf16_wmma<true,false,false,true><<<gEF, blk, 0, stream>>>(
      ln2_bf, w1_bf, bf1, nullptr, nullptr, h_bf, SEQ, DFF, EMBED);
  gemm_bf16_wmma<false,true,true,false><<<gEE, blk, 0, stream>>>(
      h_bf, w2_bf, bf2, ln2o, res3, nullptr, SEQ, EMBED, DFF);
  layernorm_k<false><<<SEQ, 256, 0, stream>>>(res3, ln3g, ln3b, out, nullptr);
}
